// ModalitySpecificPosterior_38130719653972
// MI455X (gfx1250) — compile-verified, matrix operand order checked by reading, not run on hardware
//
#include <hip/hip_runtime.h>
#include <hip/hip_bf16.h>
#include <stdint.h>

#define EPSV 1e-6f

typedef __attribute__((ext_vector_type(16))) __bf16 v16bf;
typedef __attribute__((ext_vector_type(8)))  float  v8f;

union Frag {
  uint4    q[2];
  uint32_t u[8];
  v16bf    v;
};
static_assert(sizeof(Frag) == 32, "frag size");

#if defined(__has_builtin)
#if __has_builtin(__builtin_amdgcn_cvt_pk_bf16_f32)
#define USE_CVT_PK 1
#endif
#endif

// round-to-nearest-even f32 -> bf16 in top 16 bits (returns rounded bits)
__device__ __forceinline__ uint32_t bf16r(uint32_t u) {
  return u + 0x7FFFu + ((u >> 16) & 1u);
}

// pack two f32 into one u32 of two bf16 (x -> low 16, y -> high 16)
__device__ __forceinline__ uint32_t pk_bf16(float x, float y) {
#ifdef USE_CVT_PK
  auto pk = __builtin_amdgcn_cvt_pk_bf16_f32(x, y);  // v_cvt_pk_bf16_f32
  uint32_t u;
  __builtin_memcpy(&u, &pk, 4);
  return u;
#else
  uint32_t rx = bf16r(__float_as_uint(x));
  uint32_t ry = bf16r(__float_as_uint(y));
  return (rx >> 16) | (ry & 0xFFFF0000u);
#endif
}

// split two f32 into packed bf16 hi-pair and residual lo-pair
__device__ __forceinline__ void split2(float x, float y, uint32_t& hi, uint32_t& lo) {
  uint32_t h = pk_bf16(x, y);
  float fx = __uint_as_float(h << 16);
  float fy = __uint_as_float(h & 0xFFFF0000u);
  hi = h;
  lo = pk_bf16(x - fx, y - fy);
}

__device__ __forceinline__ void cvt4(float4 v, uint2& h, uint2& l) {
  split2(v.x, v.y, h.x, l.x);
  split2(v.z, v.w, h.y, l.y);
}

// ---------------- per-row sum and sum-of-squares ----------------
__global__ void rowstats_kernel(const float* __restrict__ X, int nrows,
                                float* __restrict__ osum, float* __restrict__ osq) {
  int row  = blockIdx.x * 8 + (threadIdx.x >> 5);
  int lane = threadIdx.x & 31;
  if (row >= nrows) return;
  const float4* rp = (const float4*)(X + (size_t)row * 1024);
  float s = 0.f, ss = 0.f;
#pragma unroll
  for (int i = 0; i < 8; ++i) {
    float4 v = rp[lane + 32 * i];
    s  += v.x + v.y + v.z + v.w;
    ss += v.x * v.x + v.y * v.y + v.z * v.z + v.w * v.w;
  }
#pragma unroll
  for (int m = 1; m < 32; m <<= 1) {
    s  += __shfl_xor(s,  m, 32);
    ss += __shfl_xor(ss, m, 32);
  }
  if (lane == 0) { osum[row] = s; osq[row] = ss; }
}

// ---------------- init row accumulators ----------------
__global__ void init_kernel(float* __restrict__ se, float* __restrict__ sed,
                            int* __restrict__ mnb) {
  int i = blockIdx.x * blockDim.x + threadIdx.x;
  if (i < 8192) { se[i] = 0.f; sed[i] = 0.f; mnb[i] = 0x7F800000; } // +inf
}

// ---------------- WMMA GEMM + distance epilogue ----------------
// C tile 128x128 per 256-thread block (8 waves = 4(M) x 2(N), wave tile 32x64).
// K chunk 32. Double-buffered LDS (2x32KB = 64KB), ONE barrier per K-step.
// K-loop unrolled by 2 so buffer offsets are compile-time constants: keeps the
// backend's 16B alignment proof -> ds_load_b128 with immediate offsets.
// XOR block swizzle ((row>>2)&3 on the 4-dword block index) avoids bank
// conflicts with zero padding; it's loop-invariant per lane.
#define BUFOFS 2048  // dwords per buffer per array (128 rows * 16 dwords)

__global__ __launch_bounds__(256)
void gemm_kernel(const float* __restrict__ Q, const float* __restrict__ P,
                 const float* __restrict__ qsum, const float* __restrict__ qsq,
                 const float* __restrict__ psum, const float* __restrict__ psq,
                 float* __restrict__ se, float* __restrict__ sed, int* __restrict__ mnb,
                 float* __restrict__ post) {
  __shared__ uint32_t sAh[2 * BUFOFS], sAl[2 * BUFOFS];
  __shared__ uint32_t sBh[2 * BUFOFS], sBl[2 * BUFOFS];

  const int t    = threadIdx.x;
  const int lane = t & 31, w = t >> 5;
  const int wm = w & 3, wn = w >> 2;        // wave coords: 4 x 2
  const int r  = lane & 15, hf = lane >> 4; // fragment row / half
  const int N0 = blockIdx.x * 128, M0 = blockIdx.y * 128;

  const int lrow = t >> 3;                  // loader row 0..31 (+32*i)
  const int lcg  = t & 7;                   // loader float4 column group

  // store-side swizzled column (invariant: (row>>2)&3 == (lrow>>2)&3 for all i)
  const int sxr  = (lrow >> 2) & 3;
  const int scol = ((((lcg >> 1) ^ sxr) << 2) | ((lcg & 1) * 2));
  // read-side swizzled block columns (invariant: (rr>>2)&3 == (r>>2)&3)
  const int rxr = (r >> 2) & 3;
  const int c0  = (hf ^ rxr) << 2;          // logical block hf   (dwords 4*hf)
  const int c1  = ((hf + 2) ^ rxr) << 2;    // logical block hf+2 (dwords 8+4*hf)

  const float* qbase = Q + (size_t)(M0 + lrow) * 1024 + 4 * lcg;
  const float* pbase = P + (size_t)(N0 + lrow) * 1024 + 4 * lcg;

  const v8f zero = {0.f, 0.f, 0.f, 0.f, 0.f, 0.f, 0.f, 0.f};
  v8f acc[2][4];
#pragma unroll
  for (int mi = 0; mi < 2; ++mi)
#pragma unroll
    for (int ni = 0; ni < 4; ++ni) acc[mi][ni] = zero;

  uint2 pah[4], pal[4], pbh[4], pbl[4];     // packed staging regs
  float4 av[4], bv[4];

  auto loadAB = [&](int kk) {
#pragma unroll
    for (int i = 0; i < 4; ++i) {
      av[i] = *(const float4*)(qbase + (size_t)32768 * i + kk);
      bv[i] = *(const float4*)(pbase + (size_t)32768 * i + kk);
    }
  };
  auto cvtAB = [&]() {
#pragma unroll
    for (int i = 0; i < 4; ++i) {
      cvt4(av[i], pah[i], pal[i]);
      cvt4(bv[i], pbh[i], pbl[i]);
    }
  };
  auto storeLDS = [&](int bofs) {  // bofs is a compile-time constant at each call
#pragma unroll
    for (int i = 0; i < 4; ++i) {
      int base = bofs + (lrow + 32 * i) * 16 + scol;
      *(uint2*)&sAh[base] = pah[i];
      *(uint2*)&sAl[base] = pal[i];
      *(uint2*)&sBh[base] = pbh[i];
      *(uint2*)&sBl[base] = pbl[i];
    }
  };
  auto compute = [&](int bofs) {   // bofs is a compile-time constant at each call
    // B = P^T: lane n holds row n of P (same pattern as A fragments)
    Frag bh[4], bl[4];
#pragma unroll
    for (int ni = 0; ni < 4; ++ni) {
      int rb = bofs + (64 * wn + 16 * ni + r) * 16;
      bh[ni].q[0] = *(const uint4*)&sBh[rb + c0];
      bh[ni].q[1] = *(const uint4*)&sBh[rb + c1];
      bl[ni].q[0] = *(const uint4*)&sBl[rb + c0];
      bl[ni].q[1] = *(const uint4*)&sBl[rb + c1];
    }
#pragma unroll
    for (int mi = 0; mi < 2; ++mi) {
      Frag ah, al;
      int rb = bofs + (32 * wm + 16 * mi + r) * 16;
      ah.q[0] = *(const uint4*)&sAh[rb + c0];
      ah.q[1] = *(const uint4*)&sAh[rb + c1];
      al.q[0] = *(const uint4*)&sAl[rb + c0];
      al.q[1] = *(const uint4*)&sAl[rb + c1];
#pragma unroll
      for (int ni = 0; ni < 4; ++ni) {
        // bf16x3: qh*ph + qh*pl + ql*ph ~= f32 product
        acc[mi][ni] = __builtin_amdgcn_wmma_f32_16x16x32_bf16(
            false, ah.v, false, bh[ni].v, (short)0, acc[mi][ni], false, false);
        acc[mi][ni] = __builtin_amdgcn_wmma_f32_16x16x32_bf16(
            false, ah.v, false, bl[ni].v, (short)0, acc[mi][ni], false, false);
        acc[mi][ni] = __builtin_amdgcn_wmma_f32_16x16x32_bf16(
            false, al.v, false, bh[ni].v, (short)0, acc[mi][ni], false, false);
      }
    }
  };

  // prologue: chunk 0 -> buffer 0
  loadAB(0);
  cvtAB();
  storeLDS(0);
  __syncthreads();

  // 15 double-trips: chunks 0..29 (one barrier per K-step, constant buf offsets)
  for (int k0 = 0; k0 < 960; k0 += 64) {
    // even chunk from buffer 0; stage next into buffer 1
    loadAB(k0 + 32);
    compute(0);
    cvtAB();
    storeLDS(BUFOFS);
    __syncthreads();
    // odd chunk from buffer 1; stage next into buffer 0
    loadAB(k0 + 64);
    compute(BUFOFS);
    cvtAB();
    storeLDS(0);
    __syncthreads();
  }
  // chunk 30 from buffer 0, prefetch chunk 31 into buffer 1
  loadAB(992);
  compute(0);
  cvtAB();
  storeLDS(BUFOFS);
  __syncthreads();
  // chunk 31 from buffer 1
  compute(BUFOFS);

  // epilogue: dist = sqrt(||q||^2+||p||^2-2qp + 2eps(Sq-Sp) + D eps^2), e = exp(-dist)
  const float cD = 1024.0f * EPSV * EPSV;
#pragma unroll
  for (int mi = 0; mi < 2; ++mi) {
    const int mbase = M0 + 32 * wm + 16 * mi;
    float qs8[8], qq8[8], pse[8], psd[8], pdm[8];
#pragma unroll
    for (int j = 0; j < 8; ++j) {
      int gi = mbase + 8 * hf + j;
      qs8[j] = qsum[gi];
      qq8[j] = qsq[gi];
      pse[j] = 0.f; psd[j] = 0.f; pdm[j] = 3.0e38f;
    }
#pragma unroll
    for (int ni = 0; ni < 4; ++ni) {
      int gk = N0 + 64 * wn + 16 * ni + r;
      float pq = psq[gk], ps = psum[gk];
#pragma unroll
      for (int j = 0; j < 8; ++j) {
        int gi = mbase + 8 * hf + j;          // C layout: vgpr j -> M = j + 8*hf
        float cross = acc[mi][ni][j];
        float sq = qq8[j] + pq - 2.0f * cross + 2.0f * EPSV * (qs8[j] - ps) + cD;
        float dist = sqrtf(fmaxf(sq, 0.0f));
        float e = expf(-dist);
        post[(size_t)gi * 1024 + gk] = e;     // unnormalized; scaled in finalize
        pse[j] += e;
        psd[j] += e * dist;
        pdm[j]  = fminf(pdm[j], dist);
      }
    }
#pragma unroll
    for (int j = 0; j < 8; ++j) {
      float s = pse[j], sd = psd[j], dm = pdm[j];
#pragma unroll
      for (int m = 1; m < 16; m <<= 1) {      // reduce within each 16-lane half
        s  += __shfl_xor(s,  m, 32);
        sd += __shfl_xor(sd, m, 32);
        dm  = fminf(dm, __shfl_xor(dm, m, 32));
      }
      if (r == 0) {
        int gi = mbase + 8 * hf + j;
        atomicAdd(&se[gi],  s);
        atomicAdd(&sed[gi], sd);
        atomicMin(&mnb[gi], __float_as_int(dm)); // dist >= 0: bit order == float order
      }
    }
  }
}

// ---------------- finalize: normalize post, compute c and h ----------------
__global__ void finalize_kernel(float* __restrict__ post,
                                const float* __restrict__ se, const float* __restrict__ sed,
                                const int* __restrict__ mnb,
                                float* __restrict__ cout, float* __restrict__ hout) {
  int row = blockIdx.x;
  int t   = threadIdx.x;
  float s   = se[row];
  float inv = 1.0f / s;
  float4* rp = (float4*)(post + (size_t)row * 1024);
  float4 v = rp[t];
  v.x *= inv; v.y *= inv; v.z *= inv; v.w *= inv;
  rp[t] = v;
  if (t == 0) {
    float dmin = __int_as_float(mnb[row]);
    cout[row] = expf(-dmin) * inv;             // max_k post
    hout[row] = sed[row] * inv + logf(s);      // -sum p log p = (sum e*d)/s + log s
  }
}

extern "C" void kernel_launch(void* const* d_in, const int* in_sizes, int n_in,
                              void* d_out, int out_size, void* d_ws, size_t ws_size,
                              hipStream_t stream) {
  const float* P = (const float*)d_in[0];  // class_prototypes [1024,1024]
  const float* Q = (const float*)d_in[1];  // query_features  [8192,1024]
  float* post = (float*)d_out;             // [8192,1024]
  float* cout = post + (size_t)8192 * 1024;
  float* hout = cout + 8192;

  float* ws    = (float*)d_ws;             // ~168 KB of scratch
  float* q_sum = ws;                       // 8192
  float* q_sq  = ws + 8192;                // 8192
  float* p_sum = ws + 16384;               // 1024
  float* p_sq  = ws + 17408;               // 1024
  float* se    = ws + 18432;               // 8192
  float* sed   = ws + 26624;               // 8192
  int*   mnb   = (int*)(ws + 34816);       // 8192

  rowstats_kernel<<<1024, 256, 0, stream>>>(Q, 8192, q_sum, q_sq);
  rowstats_kernel<<<128,  256, 0, stream>>>(P, 1024, p_sum, p_sq);
  init_kernel<<<32, 256, 0, stream>>>(se, sed, mnb);
  gemm_kernel<<<dim3(8, 64), 256, 0, stream>>>(Q, P, q_sum, q_sq, p_sum, p_sq,
                                               se, sed, mnb, post);
  finalize_kernel<<<8192, 256, 0, stream>>>(post, se, sed, mnb, cout, hout);
}
